// gnn_encoder_6940667151022
// MI455X (gfx1250) — matrix operand order repaired and checked by reading, not compiled, not run on hardware
//
#include <hip/hip_runtime.h>
#include <hip/hip_bf16.h>
#include <math.h>

// Problem constants (match reference): D=128, T=8 edge types, 5 iterations.
#define D_DIM  128
#define T_TYPES 8

typedef __attribute__((ext_vector_type(16))) __bf16 bf16x16;
typedef __attribute__((ext_vector_type(8)))  float  f32x8;

union FragU { bf16x16 v; uint4 q[2]; };

__device__ __forceinline__ unsigned short f2bf(float f) {
  unsigned u = __builtin_bit_cast(unsigned, f);
  u += 0x7FFFu + ((u >> 16) & 1u);           // round-to-nearest-even
  return (unsigned short)(u >> 16);
}
__device__ __forceinline__ float bf2f(unsigned short h) {
  unsigned u = ((unsigned)h) << 16;
  return __builtin_bit_cast(float, u);
}
__device__ __forceinline__ float sigmoidf(float x) {
  return 1.0f / (1.0f + __expf(-x));
}

// A fragment (16x32 bf16, MxK) from row-major bf16 matrix with ld = 128.
// ISA 7.12.2: lane<16 -> row=lane, halves = K k0..k0+7 then k0+16..k0+23;
//             lane>=16 -> row=lane-16, halves = K k0+8..15 then k0+24..31.
__device__ __forceinline__ bf16x16 load_frag_a(const unsigned short* __restrict__ base,
                                               int row_base, int k0, int lane) {
  int l = lane & 15, sel = lane >> 4;
  const uint4* p = (const uint4*)(base + (size_t)(row_base + l) * D_DIM + k0 + sel * 8);
  FragU f;
  f.q[0] = p[0];   // k0 + sel*8 .. +7
  f.q[1] = p[2];   // k0 + sel*8 + 16 .. +23
  return f.v;
}

// B fragment from a pre-packed buffer: 32 contiguous bytes per lane.
__device__ __forceinline__ bf16x16 load_frag_b(const unsigned short* __restrict__ pk,
                                               int kt, int nt, int tiles_n, int lane) {
  const uint4* p = (const uint4*)(pk + (((size_t)kt * tiles_n + nt) * 32 + lane) * 16);
  FragU f;
  f.q[0] = p[0];
  f.q[1] = p[1];
  return f.v;
}

// ---------------------------------------------------------------------------
// One-time weight pack: f32 KxN (or transposed NxK source) -> bf16 fragments.
// B layout (dense B 32x16, matches A): lane = n + 16*h, half s<8 -> k = 8h+s, s>=8 -> k = 16+8h+(s-8).
// Packed index = (((kt*tiles_n + nt)*32 + lane)*16 + half).
// ---------------------------------------------------------------------------
__global__ void pack_b_kernel(const float* __restrict__ W, unsigned short* __restrict__ pk,
                              int Ktot, int Ntot, int transpose) {
  int total = Ktot * Ntot;
  int idx = blockIdx.x * blockDim.x + threadIdx.x;
  if (idx >= total) return;
  const float* Wb = W + (size_t)blockIdx.y * total;           // batch (edge type)
  unsigned short* pkb = pk + (size_t)blockIdx.y * total;
  int tiles_n = Ntot >> 4;
  int hh   = idx & 15;
  int lane = (idx >> 4) & 31;
  int rest = idx >> 9;
  int nt = rest % tiles_n;
  int kt = rest / tiles_n;
  int n_local = lane & 15;
  int k_local = ((hh >> 3) << 4) | ((lane >> 4) << 3) | (hh & 7);
  int kg = kt * 32 + k_local;
  int ng = nt * 16 + n_local;
  float v = transpose ? Wb[(size_t)ng * Ktot + kg] : Wb[(size_t)kg * Ntot + ng];
  pkb[idx] = f2bf(v);
}

// h = emb[x]  (writes f32 state + bf16 shadow)
__global__ void embed_kernel(const int* __restrict__ x, const float* __restrict__ emb,
                             float* __restrict__ h, unsigned short* __restrict__ hb, int N) {
  int tid = blockIdx.x * blockDim.x + threadIdx.x;
  if (tid >= N * D_DIM) return;
  int n = tid >> 7, d = tid & (D_DIM - 1);
  float v = emb[(size_t)x[n] * D_DIM + d];
  h[tid] = v;
  hb[tid] = f2bf(v);
}

// ---------------------------------------------------------------------------
// Ht[n][t][:] = hb[n][:] @ W_rel[t]   (bf16 in, f32 acc, bf16 out)
// grid = (ceil(N/128), 2, 8); block = 256 (8 waves); wave = 16 rows x 64 cols.
// ---------------------------------------------------------------------------
__global__ void rel_gemm_kernel(const unsigned short* __restrict__ hb,
                                const unsigned short* __restrict__ wrel_pk,
                                unsigned short* __restrict__ Ht, int M) {
  int wave = threadIdx.x >> 5, lane = threadIdx.x & 31;
  int row_base = (blockIdx.x * 8 + wave) * 16;
  if (row_base >= M) return;                                  // wave-uniform
  int t = blockIdx.z;
  int col_base = blockIdx.y * 64;
  const unsigned short* pk = wrel_pk + (size_t)t * D_DIM * D_DIM;

  f32x8 acc[4] = {};
#pragma unroll
  for (int kt = 0; kt < 4; ++kt) {
    bf16x16 a = load_frag_a(hb, row_base, kt * 32, lane);
#pragma unroll
    for (int c = 0; c < 4; ++c) {
      bf16x16 b = load_frag_b(pk, kt, (col_base >> 4) + c, 8, lane);
      acc[c] = __builtin_amdgcn_wmma_f32_16x16x32_bf16(
          false, a, false, b, (short)0, acc[c], false, false);
    }
  }
  int l = lane & 15, mo = (lane >> 4) * 8;
#pragma unroll
  for (int c = 0; c < 4; ++c) {
    int col = col_base + c * 16 + l;
#pragma unroll
    for (int r = 0; r < 8; ++r) {
      int row = row_base + mo + r;
      Ht[(size_t)row * (T_TYPES * D_DIM) + t * D_DIM + col] = f2bf(acc[c][r]);
    }
  }
}

// aggr[n][:] = b_node[node_type[n]][:]
__global__ void aggr_init_kernel(const int* __restrict__ node_type,
                                 const float* __restrict__ b_node,
                                 float* __restrict__ aggr, int N) {
  int tid = blockIdx.x * blockDim.x + threadIdx.x;
  if (tid >= N * D_DIM) return;
  int n = tid >> 7, d = tid & (D_DIM - 1);
  aggr[tid] = b_node[(size_t)node_type[n] * D_DIM + d];
}

// aggr[dst] += Ht[src, etype]   -- one wave per edge, 4 dims/lane, f32 atomics.
__global__ void edge_scatter_kernel(const int* __restrict__ src, const int* __restrict__ dst,
                                    const int* __restrict__ et,
                                    const unsigned short* __restrict__ Ht,
                                    float* __restrict__ aggr, int E) {
  int tid = blockIdx.x * blockDim.x + threadIdx.x;
  int e = tid >> 5;
  if (e >= E) return;
  int lane = tid & 31;
  int s = src[e], d = dst[e], t = et[e];
  const unsigned short* m = Ht + ((size_t)s * T_TYPES + t) * D_DIM + lane * 4;
  uint2 p = *(const uint2*)m;
  float* out = aggr + (size_t)d * D_DIM + lane * 4;
  unsafeAtomicAdd(out + 0, bf2f((unsigned short)(p.x & 0xFFFFu)));
  unsafeAtomicAdd(out + 1, bf2f((unsigned short)(p.x >> 16)));
  unsafeAtomicAdd(out + 2, bf2f((unsigned short)(p.y & 0xFFFFu)));
  unsafeAtomicAdd(out + 3, bf2f((unsigned short)(p.y >> 16)));
}

__global__ void cast_kernel(const float* __restrict__ in, unsigned short* __restrict__ out, int n) {
  int tid = blockIdx.x * blockDim.x + threadIdx.x;
  if (tid < n) out[tid] = f2bf(in[tid]);
}

// ---------------------------------------------------------------------------
// Fused GRU: six WMMA accumulators (i_r,i_z,i_n,h_r,h_z,h_n) per 16x16 tile,
// then gates + state update in registers. Reads hb_cur, writes h (f32, in
// place) and hb_next (separate buffer -> no cross-block RAW on A-fragments).
// grid = (ceil(N/128), 8); block = 256 (8 waves).
// ---------------------------------------------------------------------------
__global__ void gru_fused_kernel(const unsigned short* __restrict__ aggr_bf,
                                 const unsigned short* __restrict__ hb,
                                 const unsigned short* __restrict__ wih_pk,
                                 const unsigned short* __restrict__ whh_pk,
                                 const float* __restrict__ b_ih,
                                 const float* __restrict__ b_hh,
                                 float* __restrict__ h,
                                 unsigned short* __restrict__ hb_next, int M) {
  int wave = threadIdx.x >> 5, lane = threadIdx.x & 31;
  int row_base = (blockIdx.x * 8 + wave) * 16;
  if (row_base >= M) return;
  int ct = blockIdx.y;                     // column tile of D (0..7)

  f32x8 ir = {}, iz = {}, inn = {}, hr = {}, hz = {}, hn = {};
#pragma unroll
  for (int kt = 0; kt < 4; ++kt) {
    bf16x16 ai = load_frag_a(aggr_bf, row_base, kt * 32, lane);
    bf16x16 ah = load_frag_a(hb,      row_base, kt * 32, lane);
    bf16x16 b;
    b = load_frag_b(wih_pk, kt, ct,      24, lane);
    ir  = __builtin_amdgcn_wmma_f32_16x16x32_bf16(false, ai, false, b, (short)0, ir,  false, false);
    b = load_frag_b(wih_pk, kt, 8 + ct,  24, lane);
    iz  = __builtin_amdgcn_wmma_f32_16x16x32_bf16(false, ai, false, b, (short)0, iz,  false, false);
    b = load_frag_b(wih_pk, kt, 16 + ct, 24, lane);
    inn = __builtin_amdgcn_wmma_f32_16x16x32_bf16(false, ai, false, b, (short)0, inn, false, false);
    b = load_frag_b(whh_pk, kt, ct,      24, lane);
    hr  = __builtin_amdgcn_wmma_f32_16x16x32_bf16(false, ah, false, b, (short)0, hr,  false, false);
    b = load_frag_b(whh_pk, kt, 8 + ct,  24, lane);
    hz  = __builtin_amdgcn_wmma_f32_16x16x32_bf16(false, ah, false, b, (short)0, hz,  false, false);
    b = load_frag_b(whh_pk, kt, 16 + ct, 24, lane);
    hn  = __builtin_amdgcn_wmma_f32_16x16x32_bf16(false, ah, false, b, (short)0, hn,  false, false);
  }

  int l = lane & 15, mo = (lane >> 4) * 8;
  int c = ct * 16 + l;
  float bir = b_ih[c], biz = b_ih[D_DIM + c], bin = b_ih[2 * D_DIM + c];
  float bhr = b_hh[c], bhz = b_hh[D_DIM + c], bhn = b_hh[2 * D_DIM + c];
#pragma unroll
  for (int r = 0; r < 8; ++r) {
    int row = row_base + mo + r;
    float rg = sigmoidf((ir[r] + bir) + (hr[r] + bhr));
    float zg = sigmoidf((iz[r] + biz) + (hz[r] + bhz));
    float ng = tanhf((inn[r] + bin) + rg * (hn[r] + bhn));
    size_t idx = (size_t)row * D_DIM + c;
    float ho = h[idx];
    float hv = (1.0f - zg) * ng + zg * ho;
    h[idx] = hv;
    hb_next[idx] = f2bf(hv);
  }
}

extern "C" void kernel_launch(void* const* d_in, const int* in_sizes, int n_in,
                              void* d_out, int out_size, void* d_ws, size_t ws_size,
                              hipStream_t stream) {
  const int*   x          = (const int*)d_in[0];
  const int*   edge_index = (const int*)d_in[1];
  const int*   edge_type  = (const int*)d_in[2];
  const int*   node_type  = (const int*)d_in[3];
  const float* emb        = (const float*)d_in[4];
  const float* W_rel      = (const float*)d_in[5];
  const float* b_node     = (const float*)d_in[6];
  const float* W_ih       = (const float*)d_in[7];
  const float* W_hh       = (const float*)d_in[8];
  const float* b_ih       = (const float*)d_in[9];
  const float* b_hh       = (const float*)d_in[10];
  (void)n_in; (void)out_size; (void)ws_size;

  int N = in_sizes[0];
  int E = in_sizes[2];
  float* h = (float*)d_out;                 // f32 node state lives in d_out

  char* ws = (char*)d_ws;
  size_t off = 0;
  auto wsalloc = [&](size_t bytes) -> char* {
    char* p = ws + off;
    off = (off + bytes + 255) & ~(size_t)255;
    return p;
  };
  unsigned short* hb0     = (unsigned short*)wsalloc((size_t)N * D_DIM * 2);
  unsigned short* hb1     = (unsigned short*)wsalloc((size_t)N * D_DIM * 2);
  unsigned short* Ht      = (unsigned short*)wsalloc((size_t)N * T_TYPES * D_DIM * 2);
  float*          aggr    = (float*)        wsalloc((size_t)N * D_DIM * 4);
  unsigned short* agb     = (unsigned short*)wsalloc((size_t)N * D_DIM * 2);
  unsigned short* wrel_pk = (unsigned short*)wsalloc((size_t)T_TYPES * D_DIM * D_DIM * 2);
  unsigned short* wih_pk  = (unsigned short*)wsalloc((size_t)3 * D_DIM * D_DIM * 2);
  unsigned short* whh_pk  = (unsigned short*)wsalloc((size_t)3 * D_DIM * D_DIM * 2);

  // One-time weight packing into WMMA B-fragment order (bf16).
  pack_b_kernel<<<dim3((D_DIM * D_DIM + 255) / 256, T_TYPES), 256, 0, stream>>>(
      W_rel, wrel_pk, D_DIM, D_DIM, 0);
  pack_b_kernel<<<dim3((3 * D_DIM * D_DIM + 255) / 256, 1), 256, 0, stream>>>(
      W_ih, wih_pk, D_DIM, 3 * D_DIM, 1);                     // transposed read
  pack_b_kernel<<<dim3((3 * D_DIM * D_DIM + 255) / 256, 1), 256, 0, stream>>>(
      W_hh, whh_pk, D_DIM, 3 * D_DIM, 1);

  int nd = N * D_DIM;
  embed_kernel<<<(nd + 255) / 256, 256, 0, stream>>>(x, emb, h, hb0, N);

  const int* src = edge_index;
  const int* dst = edge_index + E;
  unsigned short* cur = hb0;
  unsigned short* nxt = hb1;
  int rowBlocks = (N + 127) / 128;

  for (int it = 0; it < 5; ++it) {
    rel_gemm_kernel<<<dim3(rowBlocks, D_DIM / 64, T_TYPES), 256, 0, stream>>>(
        cur, wrel_pk, Ht, N);
    aggr_init_kernel<<<(nd + 255) / 256, 256, 0, stream>>>(node_type, b_node, aggr, N);
    edge_scatter_kernel<<<(int)(((size_t)E * 32 + 255) / 256), 256, 0, stream>>>(
        src, dst, edge_type, Ht, aggr, E);
    cast_kernel<<<(nd + 255) / 256, 256, 0, stream>>>(aggr, agb, nd);
    gru_fused_kernel<<<dim3(rowBlocks, D_DIM / 16), 256, 0, stream>>>(
        agb, cur, wih_pk, whh_pk, b_ih, b_hh, h, nxt, N);
    unsigned short* t = cur; cur = nxt; nxt = t;
  }
}
